// GCN_66022237274497
// MI455X (gfx1250) — matrix operand order repaired and checked by reading, not compiled, not run on hardware
//
#include <hip/hip_runtime.h>

#define N_NODES 100000
#define N_EDGES 1600000
#define D_FEAT  512
#define N_HID   128
#define N_CLASS 40
#define N_CLS_PAD 48

typedef __attribute__((ext_vector_type(16))) __bf16 v16bf;
typedef __attribute__((ext_vector_type(8)))  __bf16 v8bf;
typedef __attribute__((ext_vector_type(8)))  float  v8f;

// ---------------- setup kernels ----------------

// rows are sorted: build CSR row_ptr via per-node lower_bound (no atomics anywhere).
__global__ void GCN_build_row_ptr(const int* __restrict__ rows, int* __restrict__ row_ptr) {
  int n = blockIdx.x * 256 + threadIdx.x;
  if (n > N_NODES) return;
  int lo = 0, hi = N_EDGES;
  while (lo < hi) { int mid = (lo + hi) >> 1; if (rows[mid] < n) lo = mid + 1; else hi = mid; }
  row_ptr[n] = lo;
}

// W0 (512x128 f32, row-major) -> W0^T bf16 (128 rows x 512 cols): B-fragment lanes
// need 16 consecutive K for a fixed N, i.e. column-major access of W0.
__global__ void GCN_build_w0t(const float* __restrict__ W0, __bf16* __restrict__ w0t) {
  int i = blockIdx.x * 256 + threadIdx.x;
  if (i >= D_FEAT * N_HID) return;
  int k = i / N_HID, n = i % N_HID;
  w0t[n * D_FEAT + k] = (__bf16)W0[i];
}

// W1 (128x40 f32) -> W1^T bf16 padded to 48 output columns (3 WMMA N-tiles).
__global__ void GCN_build_w1t(const float* __restrict__ W1, __bf16* __restrict__ w1t) {
  int i = blockIdx.x * 256 + threadIdx.x;
  if (i >= N_CLS_PAD * N_HID) return;
  int n = i / N_HID, k = i % N_HID;
  w1t[i] = (n < N_CLASS) ? (__bf16)W1[k * N_CLASS + n] : (__bf16)0.0f;
}

// ---------------- GEMM1: XW = X @ W0  (bf16 WMMA, f32 accumulate) ----------------
// One block = one 16-row M tile; 8 waves each own one 16-col N tile.
__global__ __launch_bounds__(256) void GCN_gemm_xw0(const float* __restrict__ x,
                                                    const __bf16* __restrict__ w0t,
                                                    float* __restrict__ xw) {
  __shared__ __bf16 As[16][D_FEAT + 16];   // +16 bf16 pad: kills LDS bank conflicts
  const int row0 = blockIdx.x * 16;
  // stage 16x512 f32 -> bf16 LDS tile (256 threads, 32 elems each)
  for (int i = threadIdx.x; i < 16 * D_FEAT; i += 256) {
    int r = i >> 9, c = i & (D_FEAT - 1);
    As[r][c] = (__bf16)x[(size_t)(row0 + r) * D_FEAT + c];
  }
  __syncthreads();

  const int wave = threadIdx.x >> 5;      // N tile id 0..7
  const int lane = threadIdx.x & 31;
  const int lr = lane & 15;               // row (A) / column (B) within tile
  const int lh = lane >> 4;               // lane half selects K sub-chunks
  v8f acc = {};
  #pragma unroll
  for (int k0 = 0; k0 < D_FEAT; k0 += 32) {
    // A fragment (16x32 bf16): lane-lo holds K {0..15}, lane-hi {16..31}
    v8bf a0 = *(const v8bf*)(&As[lr][k0 + lh * 16]);
    v8bf a1 = *(const v8bf*)(&As[lr][k0 + lh * 16 + 8]);
    v16bf a = __builtin_shufflevector(a0, a1, 0,1,2,3,4,5,6,7,8,9,10,11,12,13,14,15);
    // B fragment (32x16 bf16): lane holds 16 consecutive K of column N=lr
    v16bf b = *(const v16bf*)(&w0t[(size_t)(wave * 16 + lr) * D_FEAT + k0 + lh * 16]);
    acc = __builtin_amdgcn_wmma_f32_16x16x32_bf16(false, a, false, b,
                                                  (short)0, acc, false, false);
  }
  // D layout: VGPR v, lane L -> M = v + 8*(L>>4), N = L&15
  const int n = wave * 16 + lr;
  #pragma unroll
  for (int v = 0; v < 8; ++v)
    xw[(size_t)(row0 + v + 8 * lh) * N_HID + n] = acc[v];
}

// ---------------- SpMM1 + bias + ReLU: H = relu(A @ XW + b0) ----------------
// One wave per node; lane owns 4 contiguous features (float4). XW is L2-resident.
__global__ __launch_bounds__(256) void GCN_spmm_relu(const float* __restrict__ xw,
                                                     const int* __restrict__ row_ptr,
                                                     const int* __restrict__ cols,
                                                     const float* __restrict__ vals,
                                                     const float* __restrict__ b0,
                                                     float* __restrict__ H) {
  const int node = blockIdx.x * 8 + (threadIdx.x >> 5);
  if (node >= N_NODES) return;
  const int lane = threadIdx.x & 31;
  const int e0 = row_ptr[node], e1 = row_ptr[node + 1];
  const float4* __restrict__ xw4 = (const float4*)xw;   // row = 32 float4
  float4 acc = make_float4(0.f, 0.f, 0.f, 0.f);
  for (int e = e0; e < e1; ++e) {        // e is wave-uniform -> scalar loads
    const int c = cols[e];
    const float v = vals[e];
    float4 t = xw4[(size_t)c * 32 + lane];
    acc.x += v * t.x; acc.y += v * t.y; acc.z += v * t.z; acc.w += v * t.w;
  }
  float4 bb = ((const float4*)b0)[lane];
  float4 r;
  r.x = fmaxf(acc.x + bb.x, 0.f); r.y = fmaxf(acc.y + bb.y, 0.f);
  r.z = fmaxf(acc.z + bb.z, 0.f); r.w = fmaxf(acc.w + bb.w, 0.f);
  ((float4*)H)[(size_t)node * 32 + lane] = r;
}

// ---------------- GEMM2: HW = H @ W1  (bf16 WMMA, padded N=48) ----------------
__global__ __launch_bounds__(96) void GCN_gemm_hw1(const float* __restrict__ H,
                                                   const __bf16* __restrict__ w1t,
                                                   float* __restrict__ hw) {
  __shared__ __bf16 Hs[16][N_HID + 16];
  const int row0 = blockIdx.x * 16;
  for (int i = threadIdx.x; i < 16 * N_HID; i += 96) {
    int r = i >> 7, c = i & (N_HID - 1);
    Hs[r][c] = (__bf16)H[(size_t)(row0 + r) * N_HID + c];
  }
  __syncthreads();

  const int wave = threadIdx.x >> 5;      // N tile 0..2
  const int lane = threadIdx.x & 31;
  const int lr = lane & 15, lh = lane >> 4;
  v8f acc = {};
  #pragma unroll
  for (int k0 = 0; k0 < N_HID; k0 += 32) {
    v8bf a0 = *(const v8bf*)(&Hs[lr][k0 + lh * 16]);
    v8bf a1 = *(const v8bf*)(&Hs[lr][k0 + lh * 16 + 8]);
    v16bf a = __builtin_shufflevector(a0, a1, 0,1,2,3,4,5,6,7,8,9,10,11,12,13,14,15);
    v16bf b = *(const v16bf*)(&w1t[(size_t)(wave * 16 + lr) * N_HID + k0 + lh * 16]);
    acc = __builtin_amdgcn_wmma_f32_16x16x32_bf16(false, a, false, b,
                                                  (short)0, acc, false, false);
  }
  const int n = wave * 16 + lr;
  if (n < N_CLASS) {
    #pragma unroll
    for (int v = 0; v < 8; ++v)
      hw[(size_t)(row0 + v + 8 * lh) * N_CLASS + n] = acc[v];
  }
}

// ---------------- SpMM2 + bias + softmax ----------------
// Wave per node; lane owns class `lane`, lanes 0..7 also own class lane+32.
__global__ __launch_bounds__(256) void GCN_spmm_softmax(const float* __restrict__ hw,
                                                        const int* __restrict__ row_ptr,
                                                        const int* __restrict__ cols,
                                                        const float* __restrict__ vals,
                                                        const float* __restrict__ b1,
                                                        float* __restrict__ out) {
  const int node = blockIdx.x * 8 + (threadIdx.x >> 5);
  if (node >= N_NODES) return;
  const int lane = threadIdx.x & 31;
  const int e0 = row_ptr[node], e1 = row_ptr[node + 1];
  float a0 = 0.f, a1 = 0.f;
  for (int e = e0; e < e1; ++e) {
    const int c = cols[e];
    const float v = vals[e];
    a0 += v * hw[(size_t)c * N_CLASS + lane];
    if (lane < 8) a1 += v * hw[(size_t)c * N_CLASS + 32 + lane];
  }
  float l0 = a0 + b1[lane];
  float l1 = (lane < 8) ? (a1 + b1[32 + lane]) : -__builtin_inff();
  float m = fmaxf(l0, l1);
  #pragma unroll
  for (int off = 16; off; off >>= 1) m = fmaxf(m, __shfl_xor(m, off, 32));
  float x0 = __expf(l0 - m);
  float x1 = (lane < 8) ? __expf(l1 - m) : 0.f;
  float s = x0 + x1;
  #pragma unroll
  for (int off = 16; off; off >>= 1) s += __shfl_xor(s, off, 32);
  const float inv = 1.f / s;
  out[(size_t)node * N_CLASS + lane] = x0 * inv;
  if (lane < 8) out[(size_t)node * N_CLASS + 32 + lane] = x1 * inv;
}

// ---------------- host launch ----------------
extern "C" void kernel_launch(void* const* d_in, const int* in_sizes, int n_in,
                              void* d_out, int out_size, void* d_ws, size_t ws_size,
                              hipStream_t stream) {
  const float* x    = (const float*)d_in[0];
  const int*   rows = (const int*)  d_in[1];
  const int*   cols = (const int*)  d_in[2];
  const float* vals = (const float*)d_in[3];
  const float* W0   = (const float*)d_in[4];
  const float* b0   = (const float*)d_in[5];
  const float* W1   = (const float*)d_in[6];
  const float* b1   = (const float*)d_in[7];
  float* out = (float*)d_out;
  char*  ws  = (char*)d_ws;

  // workspace layout (bytes)
  const size_t OFF_ROWPTR = 0;                        // 100001 * 4  -> pad to 512K
  const size_t OFF_XW     = 524288;                   // 100000*128*4 = 51.2 MB
  const size_t OFF_H      = OFF_XW + (size_t)N_NODES * N_HID * 4;
  const size_t OFF_HW     = OFF_H  + (size_t)N_NODES * N_HID * 4;
  const size_t OFF_W0T    = OFF_HW + (size_t)N_NODES * N_CLASS * 4;
  const size_t OFF_W1T    = OFF_W0T + (size_t)N_HID * D_FEAT * 2;

  int*    row_ptr = (int*)   (ws + OFF_ROWPTR);
  float*  xw      = (float*) (ws + OFF_XW);
  float*  Hbuf    = (float*) (ws + OFF_H);
  float*  hw      = (float*) (ws + OFF_HW);
  __bf16* w0t     = (__bf16*)(ws + OFF_W0T);
  __bf16* w1t     = (__bf16*)(ws + OFF_W1T);

  GCN_build_row_ptr<<<(N_NODES + 1 + 255) / 256, 256, 0, stream>>>(rows, row_ptr);
  GCN_build_w0t<<<(D_FEAT * N_HID + 255) / 256, 256, 0, stream>>>(W0, w0t);
  GCN_build_w1t<<<(N_CLS_PAD * N_HID + 255) / 256, 256, 0, stream>>>(W1, w1t);

  GCN_gemm_xw0<<<N_NODES / 16, 256, 0, stream>>>(x, w0t, xw);
  GCN_spmm_relu<<<N_NODES / 8, 256, 0, stream>>>(xw, row_ptr, cols, vals, b0, Hbuf);
  GCN_gemm_hw1<<<N_NODES / 16, 96, 0, stream>>>(Hbuf, w1t, hw);
  GCN_spmm_softmax<<<N_NODES / 8, 256, 0, stream>>>(hw, row_ptr, cols, vals, b1, out);
}
